// Fingerprint_27367531610659
// MI455X (gfx1250) — matrix operand order repaired and checked by reading, not compiled, not run on hardware
//
#include <hip/hip_runtime.h>
#include <hip/hip_bf16.h>
#include <math.h>

// ---------------------------------------------------------------------------
// Problem constants (from the reference)
// ---------------------------------------------------------------------------
#define BB   128
#define LL   128
#define NN   6
#define FA   39
#define FB   10
#define DD   200
#define RR   3
#define TT   2
#define MBND 256
#define KCAT 64          // Acat padded K (49 real + 15 zero)
#define NEGV (-9e8f)

typedef __attribute__((ext_vector_type(4)))  float    f32x4;
typedef __attribute__((ext_vector_type(4)))  _Float16 v4h;
typedef __attribute__((ext_vector_type(8)))  _Float16 v8h;
typedef __attribute__((ext_vector_type(16))) _Float16 v16h;
typedef __attribute__((ext_vector_type(8)))  float    v8f;

enum { ACT_NONE = 0, ACT_LRELU = 1, ACT_RELU = 2, ACT_ELU = 3 };

static __device__ __forceinline__ int imin(int a, int b) { return a < b ? a : b; }

static __device__ __forceinline__ v16h cat8(v8h lo, v8h hi) {
    return __builtin_shufflevector(lo, hi,
        0, 1, 2, 3, 4, 5, 6, 7, 8, 9, 10, 11, 12, 13, 14, 15);
}

// ---------------------------------------------------------------------------
// Generic WMMA GEMM:  C[M,Nc] = act( A[M,K] @ W + rowScale[m]*bias[n] )
//   WT == false : W is row-major [K,Nc]          (torch  x @ W)
//   WT == true  : W is row-major [Nc,K]          (torch  x @ W.T), K%4==0
//   K4 == true  : K % 4 == 0 -> vector f32x4 A staging (rows 16B aligned)
// Block = 128 threads = 4 waves; block tile = 32 rows x 64 cols of C.
// Each wave owns a 16-col strip and two 16x16 sub-tiles (two WMMAs sharing
// one B fragment).  LDS tiles are DOUBLE BUFFERED: one barrier per K step.
// Staging is branch-free (clamped loads + select-to-zero, vector DS ops).
// W tile is stored TRANSPOSED in LDS so B fragments are contiguous b128s.
// M must be a multiple of 32 (16384, 98304, 128 all are).
// ---------------------------------------------------------------------------
template <int ACT, bool WT, bool K4>
__global__ __launch_bounds__(128) void gemm_wmma_kernel(
    const float* __restrict__ A, const float* __restrict__ W,
    const float* __restrict__ bias, const float* __restrict__ rowScale,
    float* __restrict__ C, int M, int K, int Nc)
{
    __shared__ __align__(32) _Float16 As[2][32][32];    // [buf][m][k]
    __shared__ __align__(32) _Float16 WsT[2][64][32];   // [buf][n][k] (transposed)

    const int tid  = threadIdx.x;
    const int wave = tid >> 5;
    const int lane = tid & 31;
    const int m0   = blockIdx.x * 32;
    const int n0   = blockIdx.y * 64;

    v8f acc0 = {}, acc1 = {};
    const int kTiles = (K + 31) >> 5;

    // ---- staging routine (branch-free, clamped addresses) ----
    auto do_stage = [&](int buf, int k0) {
        {   // ---- A tile: 32 rows x 32 k; one thread = 8 consecutive k ----
            const int row = tid >> 2;                  // 0..31
            const int seg = (tid & 3) * 8;             // 0,8,16,24
            const int k   = k0 + seg;
            const float* Ap = A + (size_t)(m0 + row) * K;
            v4h h0, h1;
            if (K4) {
                f32x4 v0 = *(const f32x4*)(Ap + imin(k, K - 4));
                f32x4 v1 = *(const f32x4*)(Ap + imin(k + 4, K - 4));
                if (k >= K)     { v0[0] = 0.f; v0[1] = 0.f; v0[2] = 0.f; v0[3] = 0.f; }
                if (k + 4 >= K) { v1[0] = 0.f; v1[1] = 0.f; v1[2] = 0.f; v1[3] = 0.f; }
                h0 = __builtin_convertvector(v0, v4h);
                h1 = __builtin_convertvector(v1, v4h);
            } else {
#pragma unroll
                for (int c = 0; c < 4; ++c) {
                    const float x0 = Ap[imin(k + c, K - 1)];
                    const float x1 = Ap[imin(k + 4 + c, K - 1)];
                    h0[c] = (_Float16)((k + c < K) ? x0 : 0.f);
                    h1[c] = (_Float16)((k + 4 + c < K) ? x1 : 0.f);
                }
            }
            *(v4h*)&As[buf][row][seg]     = h0;
            *(v4h*)&As[buf][row][seg + 4] = h1;
        }
        {   // ---- W tile transposed: WsT[n - n0][k - k0], 64 n x 32 k ----
            const int j    = tid & 63;
            const int kseg = (tid >> 6) * 16;          // 0 or 16
            const int n    = n0 + j;
            const int nc   = imin(n, Nc - 1);
            f32x4 v[4];
            if (WT) {
                const float* Wp = W + (size_t)nc * K;  // K%4==0 for WT uses
#pragma unroll
                for (int q = 0; q < 4; ++q) {
                    const int k = k0 + kseg + q * 4;
                    v[q] = *(const f32x4*)(Wp + imin(k, K - 4));
#pragma unroll
                    for (int c = 0; c < 4; ++c)
                        if (k + c >= K || n >= Nc) v[q][c] = 0.f;
                }
            } else {
#pragma unroll
                for (int q = 0; q < 4; ++q) {
#pragma unroll
                    for (int c = 0; c < 4; ++c) {
                        const int k = k0 + kseg + q * 4 + c;
                        const float x = W[(size_t)imin(k, K - 1) * Nc + nc];
                        v[q][c] = (k < K && n < Nc) ? x : 0.f;
                    }
                }
            }
#pragma unroll
            for (int q = 0; q < 4; ++q)
                *(v4h*)&WsT[buf][j][kseg + q * 4] = __builtin_convertvector(v[q], v4h);
        }
    };

    do_stage(0, 0);
    __syncthreads();

    for (int kt = 0; kt < kTiles; ++kt) {
        const int buf = kt & 1;
        if (kt + 1 < kTiles) do_stage(buf ^ 1, (kt + 1) * 32);

        // ---- fragments per gfx1250 ISA layout ----
        const int half = lane >> 4;
        const int mrow = lane & 15;
        const v16h af0 = cat8(*(const v8h*)&As[buf][mrow][half * 8],
                              *(const v8h*)&As[buf][mrow][16 + half * 8]);
        const v16h af1 = cat8(*(const v8h*)&As[buf][16 + mrow][half * 8],
                              *(const v8h*)&As[buf][16 + mrow][16 + half * 8]);
        const int bcol = wave * 16 + (lane & 15);
        const v16h bf  = cat8(*(const v8h*)&WsT[buf][bcol][half * 8],
                              *(const v8h*)&WsT[buf][bcol][16 + half * 8]);

        acc0 = __builtin_amdgcn_wmma_f32_16x16x32_f16(
            false, af0, false, bf, (short)0, acc0, false, false);
        acc1 = __builtin_amdgcn_wmma_f32_16x16x32_f16(
            false, af1, false, bf, (short)0, acc1, false, false);
        __syncthreads();
    }

    // ---- epilogue: C layout -> VGPR v: M = v + 8*(lane>>4), N = lane&15
    const int half = lane >> 4;
    const int nOut = n0 + wave * 16 + (lane & 15);
    if (nOut < Nc) {
        const float bs = bias ? bias[nOut] : 0.0f;
#pragma unroll
        for (int v = 0; v < 8; ++v) {
            const int ma = m0 + v + half * 8;
            const int mb = ma + 16;
            const float ra = rowScale ? rowScale[ma] : 1.0f;
            const float rb = rowScale ? rowScale[mb] : 1.0f;
            float xa = acc0[v] + ra * bs;
            float xb = acc1[v] + rb * bs;
            if (ACT == ACT_LRELU)      { xa = (xa > 0.f) ? xa : 0.01f * xa;
                                         xb = (xb > 0.f) ? xb : 0.01f * xb; }
            else if (ACT == ACT_RELU)  { xa = fmaxf(xa, 0.f); xb = fmaxf(xb, 0.f); }
            else if (ACT == ACT_ELU)   { xa = (xa > 0.f) ? xa : (expf(xa) - 1.0f);
                                         xb = (xb > 0.f) ? xb : (expf(xb) - 1.0f); }
            C[(size_t)ma * Nc + nOut] = xa;
            C[(size_t)mb * Nc + nOut] = xb;
        }
    }
}

// ---------------------------------------------------------------------------
// Build concatenated gathered features for round-0 neighbors, padded to K=64.
// ---------------------------------------------------------------------------
__global__ __launch_bounds__(256) void build_cat_kernel(
    const float* __restrict__ atom_list, const float* __restrict__ bond_list,
    const int* __restrict__ adeg, const int* __restrict__ bdeg,
    float* __restrict__ Acat)
{
    const int row = blockIdx.x * blockDim.x + threadIdx.x;
    if (row >= BB * LL * NN) return;
    const int b = row / (LL * NN);
    const int ai = adeg[row];
    const int bi = bdeg[row];
    const float* ap = atom_list + ((size_t)b * LL + ai) * FA;
    const float* bp = bond_list + ((size_t)b * MBND + bi) * FB;
    float* op = Acat + (size_t)row * KCAT;
#pragma unroll 4
    for (int c = 0; c < FA; ++c) op[c] = ap[c];
#pragma unroll
    for (int c = 0; c < FB; ++c) op[FA + c] = bp[c];
#pragma unroll
    for (int c = FA + FB; c < KCAT; ++c) op[c] = 0.0f;
}

// ---------------------------------------------------------------------------
// Fused neighbor attention + aggregation (one wave per atom).
// align scores -> masked 6-way softmax -> agg = sum_n w_n*nei_n, wsum = sum w.
// (weights don't depend on trans, so the 98304-row attend GEMM collapses
//  to 16384 rows: ctx = elu(agg@W_attend + wsum*b_attend).)
// ---------------------------------------------------------------------------
__global__ __launch_bounds__(256) void aggregate_kernel(
    const float* __restrict__ cur, const float* __restrict__ nei0,
    const int* __restrict__ adeg,
    const float* __restrict__ W_align, const float* __restrict__ b_align,
    float* __restrict__ agg, float* __restrict__ wsum, int round0)
{
    const int wave = threadIdx.x >> 5;
    const int lane = threadIdx.x & 31;
    const int atom = blockIdx.x * 8 + wave;            // 0 .. B*L-1
    const int b    = atom >> 7;                        // / L
    const float* curp = cur + (size_t)atom * DD;

    float selfdot = 0.f;
    for (int k = lane; k < DD; k += 32) selfdot += curp[k] * W_align[k];
#pragma unroll
    for (int off = 16; off; off >>= 1) selfdot += __shfl_xor(selfdot, off, 32);

    float score[NN];
    const float* neip[NN];
    bool pad[NN];
#pragma unroll
    for (int n = 0; n < NN; ++n) {
        const int idx = adeg[(size_t)atom * NN + n];
        pad[n] = (idx == LL - 1);
        neip[n] = round0 ? (nei0 + ((size_t)atom * NN + n) * DD)
                         : (cur + ((size_t)b * LL + idx) * DD);
        float d = 0.f;
        for (int k = lane; k < DD; k += 32) d += neip[n][k] * W_align[DD + k];
#pragma unroll
        for (int off = 16; off; off >>= 1) d += __shfl_xor(d, off, 32);
        float s = selfdot + d + b_align[0];
        s = (s > 0.f) ? s : 0.01f * s;                 // leaky relu
        score[n] = s + (pad[n] ? NEGV : 0.f);          // softmax mask
    }
    float smax = score[0];
#pragma unroll
    for (int n = 1; n < NN; ++n) smax = fmaxf(smax, score[n]);
    float sexp[NN], ssum = 0.f;
#pragma unroll
    for (int n = 0; n < NN; ++n) { sexp[n] = expf(score[n] - smax); ssum += sexp[n]; }
    const float inv = 1.f / ssum;
    float w[NN], wtot = 0.f;
#pragma unroll
    for (int n = 0; n < NN; ++n) { w[n] = pad[n] ? 0.f : sexp[n] * inv; wtot += w[n]; }
    if (lane == 0) wsum[atom] = wtot;

    for (int k = lane; k < DD; k += 32) {
        float a = 0.f;
#pragma unroll
        for (int n = 0; n < NN; ++n) a += w[n] * neip[n][k];
        agg[(size_t)atom * DD + k] = a;
    }
}

// ---------------------------------------------------------------------------
// GRU gate math (elementwise).  gi/gh are [M, 3D] in r,z,n order.
// ---------------------------------------------------------------------------
__global__ __launch_bounds__(256) void gru_gate_kernel(
    const float* __restrict__ gi, const float* __restrict__ gh,
    const float* __restrict__ hin, float* __restrict__ hout,
    float* __restrict__ actout, int M)
{
    const int i = blockIdx.x * blockDim.x + threadIdx.x;
    if (i >= M * DD) return;
    const int m = i / DD, d = i - m * DD;
    const float* gim = gi + (size_t)m * 3 * DD;
    const float* ghm = gh + (size_t)m * 3 * DD;
    const float r = 1.f / (1.f + expf(-(gim[d] + ghm[d])));
    const float z = 1.f / (1.f + expf(-(gim[DD + d] + ghm[DD + d])));
    const float n = tanhf(gim[2 * DD + d] + r * ghm[2 * DD + d]);
    const float h = (1.f - z) * n + z * hin[i];
    hout[i] = h;
    if (actout) actout[i] = fmaxf(h, 0.f);
}

// ---------------------------------------------------------------------------
// Masked molecule pooling: mol[b,:] = sum_l mask * cur[b,l,:]; act = relu.
// ---------------------------------------------------------------------------
__global__ __launch_bounds__(256) void mol_sum_kernel(
    const float* __restrict__ cur, const float* __restrict__ amask,
    float* __restrict__ mol, float* __restrict__ actm)
{
    const int b = blockIdx.x;
    for (int d = threadIdx.x; d < DD; d += blockDim.x) {
        float s = 0.f;
        for (int l = 0; l < LL; ++l)
            s += amask[b * LL + l] * cur[((size_t)b * LL + l) * DD + d];
        mol[b * DD + d] = s;
        actm[b * DD + d] = fmaxf(s, 0.f);
    }
}

// ---------------------------------------------------------------------------
// Molecule attention over atoms (block per molecule, 8 waves).
// ---------------------------------------------------------------------------
__global__ __launch_bounds__(256) void mol_attend_kernel(
    const float* __restrict__ cur, const float* __restrict__ actm,
    const float* __restrict__ amask,
    const float* __restrict__ Wma, const float* __restrict__ bma,
    float* __restrict__ aggm, float* __restrict__ wsumm)
{
    __shared__ float sc[LL];
    __shared__ float mk[LL];
    const int b = blockIdx.x;
    const int wave = threadIdx.x >> 5, lane = threadIdx.x & 31;
    const float* am = actm + (size_t)b * DD;

    float selfdot = 0.f;
    for (int k = lane; k < DD; k += 32) selfdot += am[k] * Wma[k];
#pragma unroll
    for (int off = 16; off; off >>= 1) selfdot += __shfl_xor(selfdot, off, 32);

    for (int l = wave; l < LL; l += 8) {
        const float* cp = cur + ((size_t)b * LL + l) * DD;
        float d = 0.f;
        for (int k = lane; k < DD; k += 32) d += cp[k] * Wma[DD + k];
#pragma unroll
        for (int off = 16; off; off >>= 1) d += __shfl_xor(d, off, 32);
        float s = selfdot + d + bma[0];
        s = (s > 0.f) ? s : 0.01f * s;
        if (lane == 0) {
            const float msk = amask[b * LL + l];
            sc[l] = s + ((msk == 0.f) ? NEGV : 0.f);
            mk[l] = msk;
        }
    }
    __syncthreads();
    if (threadIdx.x == 0) {               // 128-way softmax, serial is cheap
        float mx = sc[0];
        for (int l = 1; l < LL; ++l) mx = fmaxf(mx, sc[l]);
        float ssum = 0.f;
        for (int l = 0; l < LL; ++l) { sc[l] = expf(sc[l] - mx); ssum += sc[l]; }
        const float inv = 1.f / ssum;
        float wt = 0.f;
        for (int l = 0; l < LL; ++l) { sc[l] = sc[l] * inv * mk[l]; wt += sc[l]; }
        wsumm[b] = wt;
    }
    __syncthreads();
    for (int d = threadIdx.x; d < DD; d += blockDim.x) {
        float a = 0.f;
        for (int l = 0; l < LL; ++l) a += sc[l] * cur[((size_t)b * LL + l) * DD + d];
        aggm[(size_t)b * DD + d] = a;
    }
}

// feat[b] = [mol, mol + (R-2)]   (R-2 == 1, the leaked loop var)
__global__ __launch_bounds__(256) void feat_kernel(
    const float* __restrict__ mol, float* __restrict__ feat)
{
    const int i = blockIdx.x * blockDim.x + threadIdx.x;
    if (i >= BB * DD) return;
    const int b = i / DD, d = i - b * DD;
    feat[(size_t)b * 2 * DD + d]      = mol[i];
    feat[(size_t)b * 2 * DD + DD + d] = mol[i] + (float)(RR - 2);
}

// out[b] = dot(tmp[b,:], W_out[:,0]) + b_out[0]   (one wave per molecule)
__global__ __launch_bounds__(32) void final_kernel(
    const float* __restrict__ tmp, const float* __restrict__ W_out,
    const float* __restrict__ b_out, float* __restrict__ out)
{
    const int b = blockIdx.x;
    const int lane = threadIdx.x;
    float s = 0.f;
    for (int k = lane; k < DD; k += 32) s += tmp[(size_t)b * DD + k] * W_out[k];
#pragma unroll
    for (int off = 16; off; off >>= 1) s += __shfl_xor(s, off, 32);
    if (lane == 0) out[b] = s + b_out[0];
}

// ---------------------------------------------------------------------------
// Host-side orchestration
// ---------------------------------------------------------------------------
extern "C" void kernel_launch(void* const* d_in, const int* in_sizes, int n_in,
                              void* d_out, int out_size, void* d_ws, size_t ws_size,
                              hipStream_t stream) {
    (void)in_sizes; (void)n_in; (void)out_size; (void)ws_size;

    const float* atom_list = (const float*)d_in[0];
    const float* bond_list = (const float*)d_in[1];
    const int*   adeg      = (const int*)d_in[2];
    const int*   bdeg      = (const int*)d_in[3];
    const float* amask     = (const float*)d_in[4];
    const float* W_atom    = (const float*)d_in[5];
    const float* b_atom    = (const float*)d_in[6];
    const float* W_nei     = (const float*)d_in[7];
    const float* b_nei     = (const float*)d_in[8];
    const float* W_align   = (const float*)d_in[9];
    const float* b_align   = (const float*)d_in[10];
    const float* W_attend  = (const float*)d_in[11];
    const float* b_attend  = (const float*)d_in[12];
    const float* Wih       = (const float*)d_in[13];
    const float* Whh       = (const float*)d_in[14];
    const float* bih       = (const float*)d_in[15];
    const float* bhh       = (const float*)d_in[16];
    const float* W_ma      = (const float*)d_in[17];
    const float* b_ma      = (const float*)d_in[18];
    const float* W_mat     = (const float*)d_in[19];
    const float* b_mat     = (const float*)d_in[20];
    const float* mWih      = (const float*)d_in[21];
    const float* mWhh      = (const float*)d_in[22];
    const float* mbih      = (const float*)d_in[23];
    const float* mbhh      = (const float*)d_in[24];
    const float* W_metric  = (const float*)d_in[25];
    const float* b_metric  = (const float*)d_in[26];
    const float* W_out     = (const float*)d_in[27];
    const float* b_out     = (const float*)d_in[28];
    float* out = (float*)d_out;

    const int ATOMS = BB * LL;          // 16384
    const int ROWS  = BB * LL * NN;     // 98304

    // ---- workspace partition (floats) ----
    float* ws = (float*)d_ws;
    size_t o = 0;
    float* h    = ws + o; o += (size_t)ATOMS * DD;
    float* cur  = ws + o; o += (size_t)ATOMS * DD;
    float* agg  = ws + o; o += (size_t)ATOMS * DD;
    float* ctx  = ws + o; o += (size_t)ATOMS * DD;
    float* wsum = ws + o; o += (size_t)ATOMS;
    float* gi   = ws + o; o += (size_t)ATOMS * 3 * DD;
    float* gh   = ws + o; o += (size_t)ATOMS * 3 * DD;
    float* Acat = ws + o; o += (size_t)ROWS * KCAT;
    float* nei0 = ws + o; o += (size_t)ROWS * DD;
    float* mol  = ws + o; o += (size_t)BB * DD;
    float* actm = ws + o; o += (size_t)BB * DD;
    float* aggm = ws + o; o += (size_t)BB * DD;
    float* ctxm = ws + o; o += (size_t)BB * DD;
    float* wsm  = ws + o; o += (size_t)BB;
    float* gim  = ws + o; o += (size_t)BB * 3 * DD;
    float* ghm  = ws + o; o += (size_t)BB * 3 * DD;
    float* feat = ws + o; o += (size_t)BB * 2 * DD;
    float* tmpm = ws + o; o += (size_t)BB * DD;

    const dim3 blk128(128), blk256(256);

    // 1) atom_feature = lrelu(atom_list @ W_atom + b_atom)  -> h ; cur = h
    //    K = 39 (not %4) -> scalar-staging instantiation, only used here.
    gemm_wmma_kernel<ACT_LRELU, false, false>
        <<<dim3(ATOMS / 32, (DD + 63) / 64), blk128, 0, stream>>>(
        atom_list, W_atom, b_atom, nullptr, h, ATOMS, FA, DD);
    (void)hipMemcpyAsync(cur, h, (size_t)ATOMS * DD * sizeof(float),
                         hipMemcpyDeviceToDevice, stream);

    // 2) round-0 neighbor features: gather/concat (K padded to 64) then GEMM
    build_cat_kernel<<<dim3((ROWS + 255) / 256), blk256, 0, stream>>>(
        atom_list, bond_list, adeg, bdeg, Acat);
    gemm_wmma_kernel<ACT_LRELU, false, true>
        <<<dim3(ROWS / 32, (DD + 63) / 64), blk128, 0, stream>>>(
        Acat, W_nei, b_nei, nullptr, nei0, ROWS, KCAT, DD);

    // 3) R rounds of neighbor attention + GRU
    for (int r = 0; r < RR; ++r) {
        aggregate_kernel<<<dim3(ATOMS / 8), blk256, 0, stream>>>(
            cur, nei0, adeg, W_align + (size_t)r * 2 * DD, b_align + r,
            agg, wsum, (r == 0) ? 1 : 0);
        gemm_wmma_kernel<ACT_ELU, false, true>
            <<<dim3(ATOMS / 32, (DD + 63) / 64), blk128, 0, stream>>>(
            agg, W_attend + (size_t)r * DD * DD, b_attend + (size_t)r * DD, wsum,
            ctx, ATOMS, DD, DD);
        gemm_wmma_kernel<ACT_NONE, true, true>
            <<<dim3(ATOMS / 32, (3 * DD + 63) / 64), blk128, 0, stream>>>(
            ctx, Wih + (size_t)r * 3 * DD * DD, bih + (size_t)r * 3 * DD, nullptr,
            gi, ATOMS, DD, 3 * DD);
        gemm_wmma_kernel<ACT_NONE, true, true>
            <<<dim3(ATOMS / 32, (3 * DD + 63) / 64), blk128, 0, stream>>>(
            h, Whh + (size_t)r * 3 * DD * DD, bhh + (size_t)r * 3 * DD, nullptr,
            gh, ATOMS, DD, 3 * DD);
        gru_gate_kernel<<<dim3((ATOMS * DD + 255) / 256), blk256, 0, stream>>>(
            gi, gh, h, h, cur, ATOMS);
    }

    // 4) molecule pooling + T rounds of mol attention + GRU
    mol_sum_kernel<<<dim3(BB), blk256, 0, stream>>>(cur, amask, mol, actm);
    for (int t = 0; t < TT; ++t) {
        mol_attend_kernel<<<dim3(BB), blk256, 0, stream>>>(
            cur, actm, amask, W_ma, b_ma, aggm, wsm);
        gemm_wmma_kernel<ACT_ELU, false, true>
            <<<dim3(BB / 32, (DD + 63) / 64), blk128, 0, stream>>>(
            aggm, W_mat, b_mat, wsm, ctxm, BB, DD, DD);
        gemm_wmma_kernel<ACT_NONE, true, true>
            <<<dim3(BB / 32, (3 * DD + 63) / 64), blk128, 0, stream>>>(
            ctxm, mWih, mbih, nullptr, gim, BB, DD, 3 * DD);
        gemm_wmma_kernel<ACT_NONE, true, true>
            <<<dim3(BB / 32, (3 * DD + 63) / 64), blk128, 0, stream>>>(
            mol, mWhh, mbhh, nullptr, ghm, BB, DD, 3 * DD);
        gru_gate_kernel<<<dim3((BB * DD + 255) / 256), blk256, 0, stream>>>(
            gim, ghm, mol, mol, actm, BB);
    }

    // 5) readout
    feat_kernel<<<dim3((BB * DD + 255) / 256), blk256, 0, stream>>>(mol, feat);
    gemm_wmma_kernel<ACT_NONE, false, true>
        <<<dim3(BB / 32, (DD + 63) / 64), blk128, 0, stream>>>(
        feat, W_metric, b_metric, nullptr, tmpm, BB, 2 * DD, DD);
    final_kernel<<<dim3(BB), dim3(32), 0, stream>>>(tmpm, W_out, b_out, out);
}
